// MemoryQueueContrastiveLoss_8469675508139
// MI455X (gfx1250) — matrix operand order repaired and checked by reading, compiled, not run on hardware
//
#include <hip/hip_runtime.h>
#include <hip/hip_bf16.h>
#include <math.h>

#define BATCH 1024
#define DIM   128
#define QSIZE 65536
#define INIT_TEMP 0.07f
#define MAX_TEMPV (0.07f * 1.3f)

typedef __attribute__((ext_vector_type(2)))  float  v2f;
typedef __attribute__((ext_vector_type(8)))  float  v8f;
typedef __attribute__((ext_vector_type(16))) __bf16 v16bf;

union Frag16 { uint4 q[2]; v16bf v; };

__device__ __forceinline__ unsigned short f2bf(float f) {
  unsigned int u = __float_as_uint(f);
  u += 0x7FFFu + ((u >> 16) & 1u);            // round-to-nearest-even
  return (unsigned short)(u >> 16);
}

__device__ __forceinline__ void sched_scalars(int fill, float* inv_temp, float* qweight) {
  float fr = fminf((float)fill, (float)QSIZE) / (float)QSIZE;
  float t = MAX_TEMPV - (MAX_TEMPV - INIT_TEMP) * fr;
  if (fr >= 0.95f) t = INIT_TEMP;
  float w = fminf(1.0f, fr * 1.5f);
  if (fr < 0.2f) w = fr * 0.5f;
  *inv_temp = 1.0f / t;
  *qweight = w;
}

// ---------------- K0: zero accumulators ----------------
__global__ void k_zero(float* __restrict__ p, int n) {
  int i = blockIdx.x * blockDim.x + threadIdx.x;
  if (i < n) p[i] = 0.0f;
}

// ---------------- K1: L2-normalize rows, f32 + bf16 copies ----------------
__global__ void k_normalize(const float* __restrict__ vin, const float* __restrict__ tin,
                            float* __restrict__ vnf, float* __restrict__ tnf,
                            unsigned short* __restrict__ vnb, unsigned short* __restrict__ tnb) {
  int wave = (blockIdx.x * blockDim.x + threadIdx.x) >> 5;  // 0..2047, one row per wave32
  int lane = threadIdx.x & 31;
  const float* src; float* dstf; unsigned short* dstb; int row;
  if (wave < BATCH) { src = vin; dstf = vnf; dstb = vnb; row = wave; }
  else              { src = tin; dstf = tnf; dstb = tnb; row = wave - BATCH; }
  const float4 x = *(const float4*)(src + row * DIM + lane * 4);
  float s = x.x * x.x + x.y * x.y + x.z * x.z + x.w * x.w;
  #pragma unroll
  for (int m = 16; m >= 1; m >>= 1) s += __shfl_xor(s, m, 32);
  float rn = 1.0f / sqrtf(s);
  float4 y = make_float4(x.x * rn, x.y * rn, x.z * rn, x.w * rn);
  *(float4*)(dstf + row * DIM + lane * 4) = y;
  uint2 p;
  p.x = (unsigned)f2bf(y.x) | ((unsigned)f2bf(y.y) << 16);
  p.y = (unsigned)f2bf(y.z) | ((unsigned)f2bf(y.w) << 16);
  *(uint2*)(dstb + row * DIM + lane * 4) = p;
}

// ---------------- K2: sims = (vn @ tn^T) / temp, f32 WMMA 16x16x4 ----------------
__global__ void k_sims(const float* __restrict__ vnf, const float* __restrict__ tnf,
                       float* __restrict__ sims, const int* __restrict__ fill) {
  float invt, qw; sched_scalars(*fill, &invt, &qw);
  int wave = threadIdx.x >> 5;
  int lane = threadIdx.x & 31;
  int tile = blockIdx.x * 8 + wave;          // 0..4095 = 64x64 tiles of 16x16
  int ti = tile >> 6, tj = tile & 63;
  int row0 = ti * 16, col0 = tj * 16;
  int half = lane >> 4, idx = lane & 15;
  // A 16x4 f32: lanes 0-15 K={k0,k0+1}, lanes 16-31 K={k0+2,k0+3}; M = idx.
  // B 4x16 f32: same K split; N = idx; B[k][n] = tn[n][k] (contiguous along k).
  const float* ap = vnf + (row0 + idx) * DIM + half * 2;
  const float* bp = tnf + (col0 + idx) * DIM + half * 2;
  v8f acc = {};
  #pragma unroll 8
  for (int k0 = 0; k0 < DIM; k0 += 4) {
    v2f a = *(const v2f*)(ap + k0);
    v2f b = *(const v2f*)(bp + k0);
    acc = __builtin_amdgcn_wmma_f32_16x16x4_f32(false, a, false, b, (short)0, acc,
                                                false, false);
  }
  #pragma unroll
  for (int j = 0; j < 8; ++j) {
    int r = row0 + half * 8 + j;             // C: VGPR j -> M=j (lanes 0-15), M=8+j (16-31)
    sims[r * BATCH + col0 + idx] = acc[j] * invt;
  }
}

// ---------------- K3: neg[row] += sum_q exp((a_bf16 @ queue)[row,q]*qw/temp) ----------------
// queue: [DIM, QSIZE] f32 row-major; abf: [BATCH, DIM] bf16. bf16 WMMA 16x16x32.
__global__ void k_queue_neg(const float* __restrict__ queue,
                            const unsigned short* __restrict__ abf,
                            float* __restrict__ neg,
                            const int* __restrict__ fill) {
  __shared__ unsigned short Bs[16][136];     // [col][k], +8 pad to spread LDS banks
  float invt, qw; sched_scalars(*fill, &invt, &qw);
  float qs = invt * qw;

  int rg = blockIdx.x & 7;                   // 8 row groups x 128 rows
  int cg = blockIdx.x >> 3;                  // 64 col groups x 1024 cols
  int wave = threadIdx.x >> 5, lane = threadIdx.x & 31;
  int half = lane >> 4, idx = lane & 15;
  int rw = rg * 128 + wave * 16;             // this wave's 16-row band

  // A fragments (16x32 bf16, ISA layout: lanes 0-15 K {0..7,16..23}; 16-31 K {8..15,24..31})
  Frag16 afr[4];
  {
    const unsigned short* ap = abf + (size_t)(rw + idx) * DIM;
    #pragma unroll
    for (int kb = 0; kb < 4; ++kb) {
      int klo = kb * 32 + half * 8;
      afr[kb].q[0] = *(const uint4*)(ap + klo);
      afr[kb].q[1] = *(const uint4*)(ap + klo + 16);
    }
  }

  float rsum[8];
  #pragma unroll
  for (int j = 0; j < 8; ++j) rsum[j] = 0.0f;

  int scol = threadIdx.x & 15;               // staging: thread -> (col, 8 K values)
  int skg  = threadIdx.x >> 4;               // 0..15
  int sk0  = skg * 8;
  const float* gq = queue + (size_t)sk0 * QSIZE + (size_t)cg * 1024 + scol;

  for (int ct = 0; ct < 64; ++ct) {          // 64 column tiles of 16
    {
      const float* g = gq + ct * 16;
      float f[8];
      #pragma unroll
      for (int i = 0; i < 8; ++i) f[i] = g[(size_t)i * QSIZE];
      uint4 pkv;
      pkv.x = (unsigned)f2bf(f[0]) | ((unsigned)f2bf(f[1]) << 16);
      pkv.y = (unsigned)f2bf(f[2]) | ((unsigned)f2bf(f[3]) << 16);
      pkv.z = (unsigned)f2bf(f[4]) | ((unsigned)f2bf(f[5]) << 16);
      pkv.w = (unsigned)f2bf(f[6]) | ((unsigned)f2bf(f[7]) << 16);
      *(uint4*)&Bs[scol][sk0] = pkv;
      if (ct + 1 < 64) __builtin_prefetch(g + 16, 0, 0);   // global_prefetch_b8
    }
    __syncthreads();
    v8f acc = {};
    #pragma unroll
    for (int kb = 0; kb < 4; ++kb) {
      // B 32x16 bf16: lane -> N=idx; K = kb*32 + half*8 + {0..7, 16..23}
      Frag16 bfr;
      const unsigned short* bp = &Bs[idx][kb * 32 + half * 8];
      bfr.q[0] = *(const uint4*)bp;
      bfr.q[1] = *(const uint4*)(bp + 16);
      acc = __builtin_amdgcn_wmma_f32_16x16x32_bf16(false, afr[kb].v, false, bfr.v,
                                                    (short)0, acc, false, false);
    }
    __syncthreads();
    #pragma unroll
    for (int j = 0; j < 8; ++j) rsum[j] += __expf(acc[j] * qs);
  }

  // reduce each 16-lane half (columns), one f32 atomic per row
  #pragma unroll
  for (int j = 0; j < 8; ++j) {
    float s = rsum[j];
    s += __shfl_xor(s, 1, 32);
    s += __shfl_xor(s, 2, 32);
    s += __shfl_xor(s, 4, 32);
    s += __shfl_xor(s, 8, 32);
    if (idx == 0) atomicAdd(&neg[rw + half * 8 + j], s);
  }
}

// ---------------- K4: batch negatives (masked exp, row+col sums) ----------------
__global__ void k_batchneg(const float* __restrict__ sims, const int* __restrict__ mid,
                           float* __restrict__ neg_v, float* __restrict__ neg_t) {
  int i = (blockIdx.x * blockDim.x + threadIdx.x) >> 5;   // one row per wave
  int lane = threadIdx.x & 31;
  int mi = mid[i];
  float rs = 0.0f;
  for (int j0 = 0; j0 < BATCH; j0 += 32) {
    int j = j0 + lane;
    float s = sims[i * BATCH + j];
    float e = (mid[j] == mi) ? 0.0f : __expf(s);
    rs += e;
    atomicAdd(&neg_t[j], e);
  }
  #pragma unroll
  for (int m = 16; m >= 1; m >>= 1) rs += __shfl_xor(rs, m, 32);
  if (lane == 0) atomicAdd(&neg_v[i], rs);
}

// ---------------- K5: positive pairs -> loss partials ----------------
__global__ void k_pos(const float* __restrict__ sims, const int* __restrict__ mid,
                      const float* __restrict__ neg_v, const float* __restrict__ neg_t,
                      float* __restrict__ acc3) {
  int i = (blockIdx.x * blockDim.x + threadIdx.x) >> 5;
  int lane = threadIdx.x & 31;
  int mi = mid[i];
  float nvi = neg_v[i];
  float lv = 0.0f, lt = 0.0f, np = 0.0f;
  for (int j0 = 0; j0 < BATCH; j0 += 32) {
    int j = j0 + lane;
    if (mid[j] == mi) {
      float s = sims[i * BATCH + j];
      float em = __expf(-s);
      lv += log1pf(nvi * em);
      lt += log1pf(neg_t[j] * em);
      np += 1.0f;
    }
  }
  #pragma unroll
  for (int m = 16; m >= 1; m >>= 1) {
    lv += __shfl_xor(lv, m, 32);
    lt += __shfl_xor(lt, m, 32);
    np += __shfl_xor(np, m, 32);
  }
  if (lane == 0) {
    atomicAdd(&acc3[0], lv);
    atomicAdd(&acc3[1], lt);
    atomicAdd(&acc3[2], np);
  }
}

// ---------------- K6: finalize ----------------
__global__ void k_final(const float* __restrict__ acc3, float* __restrict__ out) {
  float np = acc3[2];
  out[0] = 0.5f * (acc3[0] / np + acc3[1] / np);
}

extern "C" void kernel_launch(void* const* d_in, const int* in_sizes, int n_in,
                              void* d_out, int out_size, void* d_ws, size_t ws_size,
                              hipStream_t stream) {
  const float* vfe  = (const float*)d_in[0];
  const float* tfe  = (const float*)d_in[1];
  const int*   mid  = (const int*)d_in[2];
  const float* vq   = (const float*)d_in[3];   // vision_queue [DIM, QSIZE]
  const float* tq   = (const float*)d_in[4];   // text_queue   [DIM, QSIZE]
  const int*   fill = (const int*)d_in[5];

  float* ws = (float*)d_ws;                    // needs ~5.8 MB
  float* vnf = ws;                             // 131072 f32
  float* tnf = ws + 131072;                    // 131072 f32
  unsigned short* vnb = (unsigned short*)(ws + 262144); // 131072 bf16
  unsigned short* tnb = (unsigned short*)(ws + 327680); // 131072 bf16
  float* sims  = ws + 393216;                  // 1024*1024 f32
  float* neg_v = ws + 1441792;                 // 1024
  float* neg_t = ws + 1442816;                 // 1024
  float* acc3  = ws + 1443840;                 // lv, lt, num_pos

  k_zero<<<9, 256, 0, stream>>>(neg_v, 1024 + 1024 + 3);
  k_normalize<<<256, 256, 0, stream>>>(vfe, tfe, vnf, tnf, vnb, tnb);
  k_sims<<<512, 256, 0, stream>>>(vnf, tnf, sims, fill);
  k_queue_neg<<<512, 256, 0, stream>>>(tq, vnb, neg_v, fill);  // v2q = vn @ text_queue
  k_queue_neg<<<512, 256, 0, stream>>>(vq, tnb, neg_t, fill);  // t2q = tn @ vision_queue
  k_batchneg<<<128, 256, 0, stream>>>(sims, mid, neg_v, neg_t);
  k_pos<<<128, 256, 0, stream>>>(sims, mid, neg_v, neg_t, acc3);
  k_final<<<1, 1, 0, stream>>>(acc3, (float*)d_out);
}
